// MultiAttention_38783554683260
// MI455X (gfx1250) — matrix operand order repaired and checked by reading, not compiled, not run on hardware
//
#include <hip/hip_runtime.h>
#include <hip/hip_bf16.h>

typedef __attribute__((ext_vector_type(16))) _Float16 v16h;
typedef __attribute__((ext_vector_type(8)))  float    v8f;
typedef int v4i __attribute__((vector_size(16)));

#define B_   2
#define S_   2048
#define D_   1024
#define H_   16
#define HD_  64
#define M_   (B_*S_)   // 4096
#define KT_  64        // flash key-tile width
#define LDP_ 72        // padded LDS row stride (halves): 144B, conflict-free

#if defined(__gfx1250__) && __has_builtin(__builtin_amdgcn_global_load_async_to_lds_b128) && __has_builtin(__builtin_amdgcn_s_wait_asynccnt)
#define USE_ASYNC_LDS 1
#else
#define USE_ASYNC_LDS 0
#endif

// ---------------------------------------------------------------------------
// WMMA fragment helpers (wave32, 16x16x32 f16 -> f32)
// A 16x32 (MxK): lane = M (lane&15); VGPR j holds K pair at
//   k0 = 2j + (j>=4 ? 8 : 0) + 8*(lane>=16)    [ISA 7.12.2 A table]
// B 32x16 (KxN): lane = N (lane&15); VGPR j holds K pair at
//   k0 = 2j + (j>=4 ? 8 : 0) + 8*(lane>=16)    (same K order as A)
// C/D 16x16   : lane = N (lane&15); VGPR j holds row M = j + 8*(lane>=16)
// ---------------------------------------------------------------------------

__device__ __forceinline__ v16h frag_ld_a(const _Float16* rowptr, int halfsel) {
  union { v16h h; uint4 q[2]; } f;
  const uint4* p = reinterpret_cast<const uint4*>(rowptr + halfsel * 8);
  f.q[0] = p[0];  // K = 8h + 0..7      -> elements j=0..3
  f.q[1] = p[2];  // K = 16 + 8h + 0..7 -> elements j=4..7
  return f.h;
}

__device__ __forceinline__ v16h frag_ld_b(const _Float16* rowptr, int halfsel) {
  union { v16h h; uint4 q[2]; } f;
  const uint4* p = reinterpret_cast<const uint4*>(rowptr + halfsel * 8);
  f.q[0] = p[0];  // K = 8h + 0..7
  f.q[1] = p[2];  // K = 16 + 8h + 0..7
  return f.h;
}

__device__ __forceinline__ v8f wmma_f16(v16h a, v16h b, v8f c) {
  return __builtin_amdgcn_wmma_f32_16x16x32_f16(
      false, a, false, b, (short)0, c, false, false);
}

union CFrag { v8f f; float x[8]; };

// Async global->LDS b128 copy (ASYNCcnt-tracked); sync fallback.
__device__ __forceinline__ void async_ld_b128(const _Float16* g, _Float16* l) {
#if USE_ASYNC_LDS
  __builtin_amdgcn_global_load_async_to_lds_b128(
      (__attribute__((address_space(1))) v4i*)g,
      (__attribute__((address_space(3))) v4i*)l, 0, 0);
#else
  *reinterpret_cast<uint4*>(l) = *reinterpret_cast<const uint4*>(g);
#endif
}

__device__ __forceinline__ void wait_async_and_sync() {
#if USE_ASYNC_LDS
  __builtin_amdgcn_s_wait_asynccnt(0);
#endif
  __syncthreads();
}

// ---------------------------------------------------------------------------
// f32 -> f16 conversion
// ---------------------------------------------------------------------------
__global__ void cvt_f32_f16(const float* __restrict__ in,
                            _Float16* __restrict__ out, int n) {
  int i = blockIdx.x * blockDim.x + threadIdx.x;
  int stride = gridDim.x * blockDim.x;
  for (; i < n; i += stride) out[i] = (_Float16)in[i];
}

// ---------------------------------------------------------------------------
// Tiled WMMA GEMM:  C[m,n] = sum_k A[m,k] * W[n,k]  (+bias, +epilogue)
// MODE 0: Q proj -> bias + RoPE -> Qm[B,H,S,hd] f16
// MODE 1: K proj -> bias + RoPE -> Km[B,H,S,hd] f16
// MODE 2: V proj -> bias        -> Vt[B,H,hd,S] f16 (transposed)
// MODE 3: O proj -> f32 d_out [M_, D_]
// Block 256 thr = 8 waves; block tile 64x128; wave tile 16x64.
// ---------------------------------------------------------------------------
template <int MODE>
__global__ __launch_bounds__(256) void gemm_proj(
    const _Float16* __restrict__ A, const _Float16* __restrict__ W,
    const float* __restrict__ bias, _Float16* __restrict__ out_h,
    float* __restrict__ out_f) {
  const int lane    = threadIdx.x & 31;
  const int wave    = threadIdx.x >> 5;
  const int halfsel = lane >> 4;
  const int lanen   = lane & 15;

  const int mg    = wave >> 1;
  const int nh    = wave & 1;
  const int mrow0 = blockIdx.x * 64 + mg * 16;
  const int ncol0 = blockIdx.y * 128 + nh * 64;

  CFrag acc[4];
#pragma unroll
  for (int g = 0; g < 4; ++g)
#pragma unroll
    for (int j = 0; j < 8; ++j) acc[g].x[j] = 0.0f;

  const _Float16* arow = A + (size_t)(mrow0 + lanen) * D_;
#pragma unroll 2
  for (int k0 = 0; k0 < D_; k0 += 32) {
    v16h af = frag_ld_a(arow + k0, halfsel);
#pragma unroll
    for (int g = 0; g < 4; ++g) {
      const _Float16* wrow = W + (size_t)(ncol0 + g * 16 + lanen) * D_ + k0;
      v16h bf = frag_ld_b(wrow, halfsel);
      acc[g].f = wmma_f16(af, bf, acc[g].f);
    }
  }

#pragma unroll
  for (int g = 0; g < 4; ++g) {
    const int n = ncol0 + g * 16 + lanen;
    if (MODE == 0 || MODE == 1) {
      const float bv = bias[n];
      const int d = n & (HD_ - 1);
      const int h = n >> 6;
      const float freq =
          __powf(10000.0f, -(float)(d & ~1) * (1.0f / (float)HD_));
#pragma unroll
      for (int j = 0; j < 8; ++j) {
        const int mrow = mrow0 + j + halfsel * 8;
        const int b = mrow >> 11;
        const int s = mrow & (S_ - 1);
        float val = acc[g].x[j] + bv;
        float partner = __shfl_xor(val, 1);  // lane parity == d parity
        float sn, cs;
        __sincosf((float)s * freq, &sn, &cs);
        float o = (d & 1) ? (partner * sn + val * cs)
                          : (val * cs - partner * sn);
        out_h[((size_t)(b * H_ + h) * S_ + s) * HD_ + d] = (_Float16)o;
      }
    } else if (MODE == 2) {
      const float bv = bias[n];
      const int d = n & (HD_ - 1);
      const int h = n >> 6;
#pragma unroll
      for (int j = 0; j < 8; ++j) {
        const int mrow = mrow0 + j + halfsel * 8;
        const int b = mrow >> 11;
        const int s = mrow & (S_ - 1);
        out_h[((size_t)(b * H_ + h) * HD_ + d) * S_ + s] =
            (_Float16)(acc[g].x[j] + bv);
      }
    } else {
#pragma unroll
      for (int j = 0; j < 8; ++j) {
        const int mrow = mrow0 + j + halfsel * 8;
        out_f[(size_t)mrow * D_ + n] = acc[g].x[j];
      }
    }
  }
}

// ---------------------------------------------------------------------------
// Flash attention: per (b,h), 128 queries per block (8 waves x 16 rows).
// 64-key tiles, double-buffered async K/V staging into padded LDS,
// 16 wmma per tile (8 QK^T + 8 PV), online softmax with cross-lane stats.
// ---------------------------------------------------------------------------
__global__ __launch_bounds__(256) void flash_attn(
    const _Float16* __restrict__ Qm,   // [B,H,S,hd]
    const _Float16* __restrict__ Km,   // [B,H,S,hd]
    const _Float16* __restrict__ Vt,   // [B,H,hd,S]
    _Float16* __restrict__ Amid) {     // [B,S,D]
  __shared__ _Float16 KtL[2][KT_][LDP_];   // 2 x 64 x 72 halves = 18 KB
  __shared__ _Float16 VtL[2][HD_][LDP_];   // 18 KB
  __shared__ _Float16 Pl[8][16][LDP_];     // 18 KB

  const int tid     = threadIdx.x;
  const int lane    = tid & 31;
  const int wave    = tid >> 5;
  const int halfsel = lane >> 4;
  const int lanen   = lane & 15;

  const int bh = blockIdx.y;
  const int b  = bh >> 4;
  const int h  = bh & (H_ - 1);
  const int q0 = blockIdx.x * 128 + wave * 16;

  const _Float16* Qbh = Qm + (size_t)bh * S_ * HD_;
  const _Float16* Kbh = Km + (size_t)bh * S_ * HD_;
  const _Float16* Vbh = Vt + (size_t)bh * HD_ * S_;

  // Q fragments, pre-scaled by 1/sqrt(D) (reference scales by input dim!)
  v16h aq0 = frag_ld_a(Qbh + (size_t)(q0 + lanen) * HD_ + 0, halfsel);
  v16h aq1 = frag_ld_a(Qbh + (size_t)(q0 + lanen) * HD_ + 32, halfsel);
  aq0 = aq0 * (_Float16)0.03125f;
  aq1 = aq1 * (_Float16)0.03125f;

  CFrag O[4];
#pragma unroll
  for (int g = 0; g < 4; ++g)
#pragma unroll
    for (int j = 0; j < 8; ++j) O[g].x[j] = 0.0f;
  float mstat[8], lstat[8];
#pragma unroll
  for (int j = 0; j < 8; ++j) { mstat[j] = -1e30f; lstat[j] = 0.0f; }

  // Stage key-tile kt into LDS buffer `buf`: K 64x64 + V 64x64 f16.
  auto stage = [&](int buf, int kt) {
    // 512 16B-chunks each; each thread issues 2+2 async b128 loads
#pragma unroll
    for (int r = 0; r < 2; ++r) {
      int c = tid + r * 256;
      int row = c >> 3, col = (c & 7) * 8;
      async_ld_b128(Kbh + (size_t)(kt + row) * HD_ + col, &KtL[buf][row][col]);
    }
#pragma unroll
    for (int r = 0; r < 2; ++r) {
      int c = tid + r * 256;
      int row = c >> 3, col = (c & 7) * 8;
      async_ld_b128(Vbh + (size_t)row * S_ + kt + col, &VtL[buf][row][col]);
    }
  };

  stage(0, 0);
  wait_async_and_sync();

  int buf = 0;
  for (int t = 0; t < S_ / KT_; ++t) {
    if (t + 1 < S_ / KT_) stage(buf ^ 1, (t + 1) * KT_);

    // ---- QK^T: S[16 x 64] = q * K_tile^T -------------------------------
    CFrag sg[4];
#pragma unroll
    for (int ng = 0; ng < 4; ++ng) {
#pragma unroll
      for (int j = 0; j < 8; ++j) sg[ng].x[j] = 0.0f;
      const _Float16* krow = &KtL[buf][ng * 16 + lanen][0];
      v16h b0 = frag_ld_b(krow, halfsel);
      v16h b1 = frag_ld_b(krow + 32, halfsel);
      sg[ng].f = wmma_f16(aq0, b0, sg[ng].f);
      sg[ng].f = wmma_f16(aq1, b1, sg[ng].f);
    }

    // ---- online softmax (row = VGPR j + 8*halfsel) ---------------------
#pragma unroll
    for (int j = 0; j < 8; ++j) {
      float v0 = sg[0].x[j], v1 = sg[1].x[j];
      float v2 = sg[2].x[j], v3 = sg[3].x[j];
      float tmax = fmaxf(fmaxf(v0, v1), fmaxf(v2, v3));
      tmax = fmaxf(tmax, __shfl_xor(tmax, 1));
      tmax = fmaxf(tmax, __shfl_xor(tmax, 2));
      tmax = fmaxf(tmax, __shfl_xor(tmax, 4));
      tmax = fmaxf(tmax, __shfl_xor(tmax, 8));
      float newm = fmaxf(mstat[j], tmax);
      float corr = __expf(mstat[j] - newm);
      float p0 = __expf(v0 - newm);
      float p1 = __expf(v1 - newm);
      float p2 = __expf(v2 - newm);
      float p3 = __expf(v3 - newm);
      float rs = (p0 + p1) + (p2 + p3);
      rs += __shfl_xor(rs, 1);
      rs += __shfl_xor(rs, 2);
      rs += __shfl_xor(rs, 4);
      rs += __shfl_xor(rs, 8);
      lstat[j] = lstat[j] * corr + rs;
      mstat[j] = newm;
#pragma unroll
      for (int g = 0; g < 4; ++g) O[g].x[j] *= corr;
      const int prow = j + halfsel * 8;
      Pl[wave][prow][lanen]      = (_Float16)p0;
      Pl[wave][prow][16 + lanen] = (_Float16)p1;
      Pl[wave][prow][32 + lanen] = (_Float16)p2;
      Pl[wave][prow][48 + lanen] = (_Float16)p3;
    }

    // ---- P * V (P via LDS C->A layout bounce) --------------------------
    v16h ap0 = frag_ld_a(&Pl[wave][lanen][0], halfsel);
    v16h ap1 = frag_ld_a(&Pl[wave][lanen][32], halfsel);
#pragma unroll
    for (int dg = 0; dg < 4; ++dg) {
      const _Float16* vrow = &VtL[buf][dg * 16 + lanen][0];
      v16h bv0 = frag_ld_b(vrow, halfsel);
      v16h bv1 = frag_ld_b(vrow + 32, halfsel);
      O[dg].f = wmma_f16(ap0, bv0, O[dg].f);
      O[dg].f = wmma_f16(ap1, bv1, O[dg].f);
    }

    wait_async_and_sync();
    buf ^= 1;
  }

#pragma unroll
  for (int g = 0; g < 4; ++g) {
    const int d = g * 16 + lanen;
#pragma unroll
    for (int j = 0; j < 8; ++j) {
      const int s = q0 + j + halfsel * 8;
      float o = O[g].x[j] / lstat[j];
      Amid[(size_t)(b * S_ + s) * D_ + h * HD_ + d] = (_Float16)o;
    }
  }
}

// ---------------------------------------------------------------------------
extern "C" void kernel_launch(void* const* d_in, const int* in_sizes, int n_in,
                              void* d_out, int out_size, void* d_ws,
                              size_t ws_size, hipStream_t stream) {
  (void)in_sizes; (void)n_in; (void)out_size; (void)ws_size;
  const float* x  = (const float*)d_in[0];
  const float* Wq = (const float*)d_in[1];
  const float* bq = (const float*)d_in[2];
  const float* Wk = (const float*)d_in[3];
  const float* bk = (const float*)d_in[4];
  const float* Wv = (const float*)d_in[5];
  const float* bv = (const float*)d_in[6];
  const float* Wo = (const float*)d_in[7];
  float* out = (float*)d_out;

  char* ws = (char*)d_ws;
  _Float16* Xh  = (_Float16*)(ws + ((size_t)0  << 20));  // [M_,D_]
  _Float16* Whq = (_Float16*)(ws + ((size_t)8  << 20));  // [D_,D_]
  _Float16* Whk = (_Float16*)(ws + ((size_t)10 << 20));
  _Float16* Whv = (_Float16*)(ws + ((size_t)12 << 20));
  _Float16* Who = (_Float16*)(ws + ((size_t)14 << 20));
  _Float16* Qm  = (_Float16*)(ws + ((size_t)16 << 20));  // [B,H,S,hd]
  _Float16* Km  = (_Float16*)(ws + ((size_t)24 << 20));
  _Float16* Vt  = (_Float16*)(ws + ((size_t)32 << 20));  // [B,H,hd,S]
  _Float16* Am  = (_Float16*)(ws + ((size_t)40 << 20));  // [B,S,D]

  const int nx = M_ * D_;
  const int nw = D_ * D_;
  cvt_f32_f16<<<dim3(512), dim3(256), 0, stream>>>(x, Xh, nx);
  cvt_f32_f16<<<dim3(256), dim3(256), 0, stream>>>(Wq, Whq, nw);
  cvt_f32_f16<<<dim3(256), dim3(256), 0, stream>>>(Wk, Whk, nw);
  cvt_f32_f16<<<dim3(256), dim3(256), 0, stream>>>(Wv, Whv, nw);
  cvt_f32_f16<<<dim3(256), dim3(256), 0, stream>>>(Wo, Who, nw);

  dim3 ggrid(M_ / 64, D_ / 128);  // (64, 8)
  gemm_proj<0><<<ggrid, 256, 0, stream>>>(Xh, Whq, bq, Qm, nullptr);
  gemm_proj<1><<<ggrid, 256, 0, stream>>>(Xh, Whk, bk, Km, nullptr);
  gemm_proj<2><<<ggrid, 256, 0, stream>>>(Xh, Whv, bv, Vt, nullptr);

  flash_attn<<<dim3(S_ / 128, B_ * H_), 256, 0, stream>>>(Qm, Km, Vt, Am);

  gemm_proj<3><<<ggrid, 256, 0, stream>>>(Am, Who, nullptr, nullptr, out);
}
